// WordEmbedding_38792144617502
// MI455X (gfx1250) — matrix operand order repaired and checked by reading, not compiled, not run on hardware
//
#include <hip/hip_runtime.h>
#include <hip/hip_bf16.h>
#include <math.h>

#define Bn    8192
#define LMAXn 16
#define MLn   10
#define EDn   300
#define EDP   320      // ED padded to multiple of 32 for WMMA K
#define CVn   256
#define Hn    256
#define G4H   1024     // 4*H
#define OUTn  512
#define N1    (Bn * MLn)   // 81920 rows, time-major (row = t*B + b)

typedef __attribute__((ext_vector_type(16))) _Float16 v16h;
typedef __attribute__((ext_vector_type(8)))  float    v8f;

// ---------------- WMMA fragment helpers (CDNA5 wave32 layouts) ----------------

// A fragment 16x32 (MxK) in this code's k order (k_pack_b matches): lanes 0-15 -> M=lane, VGPR v holds
// K = half*16 + v*2, +1   (half = lane>=16)
// Per lane-half this is one contiguous 32-byte run.
__device__ __forceinline__ v16h load_a_frag(const _Float16* __restrict__ A, int lda) {
  const int lane = threadIdx.x & 31;
  const int half = lane >> 4;
  const int m    = lane & 15;
  const _Float16* p = A + (size_t)m * lda;
  v16h a;
#pragma unroll
  for (int v = 0; v < 8; ++v) {
    const int k = (half << 4) + (v << 1);
    a[2 * v]     = p[k];
    a[2 * v + 1] = p[k + 1];
  }
  return a;
}

// Packed-B fragment: weights pre-swizzled so lane's 16 halves are contiguous.
// Tile (k0/32, n0/16) lives at tile_index*512; lane takes [lane*16, lane*16+16).
__device__ __forceinline__ v16h load_b_packed(const _Float16* __restrict__ Bp,
                                              int ktile, int ntile, int ntiles) {
  const int lane = threadIdx.x & 31;
  const _Float16* p = Bp + ((size_t)ktile * ntiles + ntile) * 512 + lane * 16;
  return *(const v16h*)p;
}

__device__ __forceinline__ v8f wmma_f16(v16h a, v16h b, v8f c) {
  return __builtin_amdgcn_wmma_f32_16x16x32_f16(false, a, false, b, (short)0, c,
                                                false, false);
}

// ---------------- weight prep: fp32 -> f16 row-major, then fragment-pack ----------------

__global__ void k_conv_encw(const float* __restrict__ W, _Float16* __restrict__ O) {
  int idx = blockIdx.x * blockDim.x + threadIdx.x;
  if (idx >= EDP * CVn) return;
  int k = idx / CVn, n = idx % CVn;
  O[idx] = (k < EDn) ? (_Float16)W[k * CVn + n] : (_Float16)0.f;
}

// Wcomb[k][n] (k<256: Wih^T, else Whh^T); weights stored [N=4H, K=256] row-major
__global__ void k_conv_comb(const float* __restrict__ Wih,
                            const float* __restrict__ Whh,
                            _Float16* __restrict__ O) {
  int idx = blockIdx.x * blockDim.x + threadIdx.x;
  if (idx >= 512 * G4H) return;
  int k = idx / G4H, n = idx % G4H;
  float v = (k < CVn) ? Wih[n * CVn + k] : Whh[n * Hn + (k - CVn)];
  O[idx] = (_Float16)v;
}

__global__ void k_conv_outw(const float* __restrict__ W, _Float16* __restrict__ O) {
  int idx = blockIdx.x * blockDim.x + threadIdx.x;
  if (idx >= 512 * OUTn) return;
  O[idx] = (_Float16)W[idx];
}

// Pack row-major f16 [K,N] into WMMA B-fragment order.
// packed[tile*512 + lane*16 + 2v + s] =
//   src[(ktile*32 + (lane>>4)*16 + 2v + s) * N + ntile*16 + (lane&15)]
__global__ void k_pack_b(const _Float16* __restrict__ S, _Float16* __restrict__ P,
                         int K, int N) {
  int idx = blockIdx.x * blockDim.x + threadIdx.x;
  if (idx >= K * N) return;
  int ntiles = N >> 4;
  int tile   = idx >> 9;          // /512
  int within = idx & 511;
  int lane   = within >> 4;
  int v2     = within & 15;       // 2v + s
  int ktile  = tile / ntiles;
  int ntile  = tile % ntiles;
  int n      = (ntile << 4) + (lane & 15);
  int k      = (ktile << 5) + ((lane >> 4) << 4) + v2;
  P[idx] = S[(size_t)k * N + n];
}

__global__ void k_bias2(const float* __restrict__ a, const float* __restrict__ b,
                        float* __restrict__ o, int n) {
  int idx = blockIdx.x * blockDim.x + threadIdx.x;
  if (idx < n) o[idx] = a[idx] + b[idx];
}

// ---------------- stage 1: embedding gather + bilinear length resize ----------------

__global__ void k_embed(const int* __restrict__ words, const int* __restrict__ lengths,
                        const float* __restrict__ emb, _Float16* __restrict__ X) {
  const long total = (long)N1 * EDP;
  for (long idx = blockIdx.x * (long)blockDim.x + threadIdx.x; idx < total;
       idx += (long)gridDim.x * blockDim.x) {
    int d = (int)(idx % EDP);
    int r = (int)(idx / EDP);        // r = t*B + b  (time-major)
    int t = r / Bn, b = r % Bn;
    _Float16 o = (_Float16)0.f;
    if (d < EDn) {
      int   L   = lengths[b];
      float pos = (float)t * (float)(L - 1) / (float)(MLn - 1);
      int   i0  = (int)floorf(pos);
      int   i1  = i0 + 1; if (i1 > L - 1) i1 = L - 1;
      float w   = pos - (float)i0;
      int c0 = words[b * LMAXn + i0];
      int c1 = words[b * LMAXn + i1];
      float val = (1.f - w) * emb[c0 * EDn + d] + w * emb[c1 * EDn + d];
      o = (_Float16)val;
    }
    X[idx] = o;
  }
}

// Core MMA step: all 6 fragment loads hoisted before the 8 WMMAs so the loads issue as
// one clause and the WMMAs retire back-to-back.
#define MMA_STEP(A0PTR, A1PTR, LDA, WP, KT, CT, NT)                       \
  do {                                                                    \
    v16h a0 = load_a_frag((A0PTR), (LDA));                                \
    v16h a1 = load_a_frag((A1PTR), (LDA));                                \
    v16h b0 = load_b_packed((WP), (KT), (CT) + 0, (NT));                  \
    v16h b1 = load_b_packed((WP), (KT), (CT) + 1, (NT));                  \
    v16h b2 = load_b_packed((WP), (KT), (CT) + 2, (NT));                  \
    v16h b3 = load_b_packed((WP), (KT), (CT) + 3, (NT));                  \
    acc[0][0] = wmma_f16(a0, b0, acc[0][0]);                              \
    acc[1][0] = wmma_f16(a1, b0, acc[1][0]);                              \
    acc[0][1] = wmma_f16(a0, b1, acc[0][1]);                              \
    acc[1][1] = wmma_f16(a1, b1, acc[1][1]);                              \
    acc[0][2] = wmma_f16(a0, b2, acc[0][2]);                              \
    acc[1][2] = wmma_f16(a1, b2, acc[1][2]);                              \
    acc[0][3] = wmma_f16(a0, b3, acc[0][3]);                              \
    acc[1][3] = wmma_f16(a1, b3, acc[1][3]);                              \
  } while (0)

// ---------------- stage 2: encoder GEMM [N1,320]x[320,256] + bias + ReLU ----------------
// Wave tile: 32 (M) x 64 (N). Block = 4 waves stacked in M -> 128 x 64 block tile.

__global__ void k_gemm_enc(const _Float16* __restrict__ X, const _Float16* __restrict__ Wp,
                           const float* __restrict__ bias, _Float16* __restrict__ Y) {
  const int wid  = threadIdx.x >> 5;
  const int row0 = blockIdx.x * 128 + wid * 32;
  const int col0 = blockIdx.y * 64;
  const int ntiles = CVn >> 4;
  v8f z = 0;
  v8f acc[2][4] = {{z, z, z, z}, {z, z, z, z}};
  for (int k0 = 0; k0 < EDP; k0 += 32) {
    MMA_STEP(X + (size_t)row0 * EDP + k0, X + (size_t)(row0 + 16) * EDP + k0, EDP,
             Wp, k0 >> 5, col0 >> 4, ntiles);
  }
  const int lane = threadIdx.x & 31;
  const int n = lane & 15, half = lane >> 4;
#pragma unroll
  for (int mi = 0; mi < 2; ++mi)
#pragma unroll
    for (int j = 0; j < 4; ++j)
#pragma unroll
      for (int v = 0; v < 8; ++v) {
        int   r   = row0 + mi * 16 + v + half * 8;
        int   c   = col0 + 16 * j + n;
        float val = acc[mi][j][v] + bias[c];
        val = val > 0.f ? val : 0.f;
        Y[(size_t)r * CVn + c] = (_Float16)val;
      }
}

// ---------------- stage 3a: per-step gate GEMM  G = [x_t || h_prev] @ Wcomb ----------------
// kend = 256 for the first step (h_prev == 0), else 512.

__global__ void k_gemm_gate(const _Float16* __restrict__ Xt, const _Float16* __restrict__ Hp,
                            const _Float16* __restrict__ Wp, float* __restrict__ G,
                            int kend) {
  const int wid  = threadIdx.x >> 5;
  const int row0 = blockIdx.x * 128 + wid * 32;
  const int col0 = blockIdx.y * 64;
  const int ntiles = G4H >> 4;
  v8f z = 0;
  v8f acc[2][4] = {{z, z, z, z}, {z, z, z, z}};
  for (int k0 = 0; k0 < kend; k0 += 32) {
    const _Float16* A0 = (k0 < CVn) ? (Xt + (size_t)row0 * CVn + k0)
                                    : (Hp + (size_t)row0 * Hn + (k0 - CVn));
    const _Float16* A1 = (k0 < CVn) ? (Xt + (size_t)(row0 + 16) * CVn + k0)
                                    : (Hp + (size_t)(row0 + 16) * Hn + (k0 - CVn));
    // pull next K-chunk of A toward the WGP while the matrix pipe works
    __builtin_prefetch(A0 + 32, 0, 3);
    __builtin_prefetch(A1 + 32, 0, 3);
    MMA_STEP(A0, A1, CVn, Wp, k0 >> 5, col0 >> 4, ntiles);
  }
  const int lane = threadIdx.x & 31;
  const int n = lane & 15, half = lane >> 4;
#pragma unroll
  for (int mi = 0; mi < 2; ++mi)
#pragma unroll
    for (int j = 0; j < 4; ++j)
#pragma unroll
      for (int v = 0; v < 8; ++v)
        G[(size_t)(row0 + mi * 16 + v + half * 8) * G4H + col0 + 16 * j + n] =
            acc[mi][j][v];
}

// ---------------- stage 3b: LSTM gate pointwise (i,f,g,o order) ----------------

__global__ void k_gate(const float* __restrict__ G, const float* __restrict__ bias,
                       float* __restrict__ C, _Float16* __restrict__ Hout,
                       _Float16* __restrict__ Hcat, int t, int dirOff, int first) {
  int idx = blockIdx.x * blockDim.x + threadIdx.x;
  if (idx >= Bn * Hn) return;
  int b = idx >> 8, j = idx & 255;
  const float* g = G + (size_t)b * G4H;
  float gi = g[j]       + bias[j];
  float gf = g[256 + j] + bias[256 + j];
  float gg = g[512 + j] + bias[512 + j];
  float go = g[768 + j] + bias[768 + j];
  float si = 1.f / (1.f + __expf(-gi));
  float sf = 1.f / (1.f + __expf(-gf));
  float so = 1.f / (1.f + __expf(-go));
  float cp = first ? 0.f : C[idx];
  float cn = sf * cp + si * tanhf(gg);
  float h  = so * tanhf(cn);
  C[idx]    = cn;
  Hout[idx] = (_Float16)h;
  Hcat[((size_t)t * Bn + b) * (2 * Hn) + dirOff + j] = (_Float16)h;
}

// ---------------- stage 4: output GEMM [N1,512]x[512,512], scatter to [B,T,OUT] ----------------

__global__ void k_gemm_out(const _Float16* __restrict__ Hc, const _Float16* __restrict__ Wp,
                           const float* __restrict__ bias, float* __restrict__ O) {
  const int wid  = threadIdx.x >> 5;
  const int row0 = blockIdx.x * 128 + wid * 32;
  const int col0 = blockIdx.y * 64;
  const int ntiles = OUTn >> 4;
  v8f z = 0;
  v8f acc[2][4] = {{z, z, z, z}, {z, z, z, z}};
  for (int k0 = 0; k0 < 2 * Hn; k0 += 32) {
    MMA_STEP(Hc + (size_t)row0 * (2 * Hn) + k0,
             Hc + (size_t)(row0 + 16) * (2 * Hn) + k0, 2 * Hn,
             Wp, k0 >> 5, col0 >> 4, ntiles);
  }
  const int lane = threadIdx.x & 31;
  const int n = lane & 15, half = lane >> 4;
#pragma unroll
  for (int mi = 0; mi < 2; ++mi)
#pragma unroll
    for (int j = 0; j < 4; ++j)
#pragma unroll
      for (int v = 0; v < 8; ++v) {
        int r = row0 + mi * 16 + v + half * 8;   // r = t*B + b
        int t = r >> 13;                         // B = 8192 = 2^13
        int b = r & (Bn - 1);
        int c = col0 + 16 * j + n;
        O[((size_t)b * MLn + t) * OUTn + c] = acc[mi][j][v] + bias[c];
      }
}

// ---------------- host-side launch ----------------

extern "C" void kernel_launch(void* const* d_in, const int* in_sizes, int n_in,
                              void* d_out, int out_size, void* d_ws, size_t ws_size,
                              hipStream_t stream) {
  (void)in_sizes; (void)n_in; (void)out_size; (void)ws_size;
  const int*   words   = (const int*)d_in[0];
  const int*   lengths = (const int*)d_in[1];
  const float* emb     = (const float*)d_in[2];
  const float* encW    = (const float*)d_in[3];
  const float* encB    = (const float*)d_in[4];
  const float* WihF    = (const float*)d_in[5];
  const float* WhhF    = (const float*)d_in[6];
  const float* bihF    = (const float*)d_in[7];
  const float* bhhF    = (const float*)d_in[8];
  const float* WihB    = (const float*)d_in[9];
  const float* WhhB    = (const float*)d_in[10];
  const float* bihB    = (const float*)d_in[11];
  const float* bhhB    = (const float*)d_in[12];
  const float* outW    = (const float*)d_in[13];
  const float* outB    = (const float*)d_in[14];
  float* out = (float*)d_out;

  char* p = (char*)d_ws;
  auto alloc = [&](size_t bytes) {
    void* r = (void*)p;
    p += (bytes + 255) & ~(size_t)255;
    return r;
  };
  _Float16* xh     = (_Float16*)alloc((size_t)N1 * EDP * 2);
  _Float16* ench   = (_Float16*)alloc((size_t)N1 * CVn * 2);
  _Float16* encWh  = (_Float16*)alloc((size_t)EDP * CVn * 2);
  _Float16* encWp  = (_Float16*)alloc((size_t)EDP * CVn * 2);
  _Float16* wcF    = (_Float16*)alloc((size_t)512 * G4H * 2);
  _Float16* wcFp   = (_Float16*)alloc((size_t)512 * G4H * 2);
  _Float16* wcB    = (_Float16*)alloc((size_t)512 * G4H * 2);
  _Float16* wcBp   = (_Float16*)alloc((size_t)512 * G4H * 2);
  _Float16* outWh  = (_Float16*)alloc((size_t)512 * OUTn * 2);
  _Float16* outWp  = (_Float16*)alloc((size_t)512 * OUTn * 2);
  float*    bgF    = (float*)alloc((size_t)G4H * 4);
  float*    bgB    = (float*)alloc((size_t)G4H * 4);
  float*    Gbuf   = (float*)alloc((size_t)Bn * G4H * 4);
  float*    Cbuf   = (float*)alloc((size_t)Bn * Hn * 4);
  _Float16* h0     = (_Float16*)alloc((size_t)Bn * Hn * 2);
  _Float16* h1     = (_Float16*)alloc((size_t)Bn * Hn * 2);
  _Float16* hcat   = (_Float16*)alloc((size_t)N1 * 2 * Hn * 2);

  // weight prep (tiny): fp32 -> f16 row-major, then fragment-pack
  k_conv_encw<<<(EDP * CVn + 255) / 256, 256, 0, stream>>>(encW, encWh);
  k_conv_comb<<<(512 * G4H + 255) / 256, 256, 0, stream>>>(WihF, WhhF, wcF);
  k_conv_comb<<<(512 * G4H + 255) / 256, 256, 0, stream>>>(WihB, WhhB, wcB);
  k_conv_outw<<<(512 * OUTn + 255) / 256, 256, 0, stream>>>(outW, outWh);
  k_pack_b<<<(EDP * CVn + 255) / 256, 256, 0, stream>>>(encWh, encWp, EDP, CVn);
  k_pack_b<<<(512 * G4H + 255) / 256, 256, 0, stream>>>(wcF, wcFp, 512, G4H);
  k_pack_b<<<(512 * G4H + 255) / 256, 256, 0, stream>>>(wcB, wcBp, 512, G4H);
  k_pack_b<<<(512 * OUTn + 255) / 256, 256, 0, stream>>>(outWh, outWp, 512, OUTn);
  k_bias2<<<(G4H + 255) / 256, 256, 0, stream>>>(bihF, bhhF, bgF, G4H);
  k_bias2<<<(G4H + 255) / 256, 256, 0, stream>>>(bihB, bhhB, bgB, G4H);

  // embedding + resize (time-major rows)
  k_embed<<<4096, 256, 0, stream>>>(words, lengths, emb, xh);

  // encoder GEMM
  k_gemm_enc<<<dim3(N1 / 128, CVn / 64), 128, 0, stream>>>(xh, encWp, encB, ench);

  // BiLSTM: forward then backward over T=10
  for (int dir = 0; dir < 2; ++dir) {
    const _Float16* Wp = dir ? wcBp : wcFp;
    const float*    bg = dir ? bgB : bgF;
    int dirOff = dir ? Hn : 0;
    for (int s = 0; s < MLn; ++s) {
      int t = dir ? (MLn - 1 - s) : s;
      _Float16* hin  = (s & 1) ? h1 : h0;
      _Float16* hout = (s & 1) ? h0 : h1;
      int kend = (s == 0) ? CVn : 2 * Hn;   // first step: h_prev == 0
      k_gemm_gate<<<dim3(Bn / 128, G4H / 64), 128, 0, stream>>>(
          ench + (size_t)t * Bn * CVn, hin, Wp, Gbuf, kend);
      k_gate<<<(Bn * Hn + 255) / 256, 256, 0, stream>>>(
          Gbuf, bg, Cbuf, hout, hcat, t, dirOff, s == 0 ? 1 : 0);
    }
  }

  // output projection, scattered into [B, ML, OUT]
  k_gemm_out<<<dim3(N1 / 128, OUTn / 64), 128, 0, stream>>>(hcat, outWp, outB, out);
}
